// SwinTransformerBlock_14843406975530
// MI455X (gfx1250) — compile-verified, matrix operand order checked by reading, not run on hardware
//
#include <hip/hip_runtime.h>
#include <hip/hip_bf16.h>
#include <math.h>

// ---------------------------------------------------------------------------
// Swin block on gfx1250 (MI455X): all GEMMs + attention via
// v_wmma_f32_16x16x32_f16, f16 intermediate activations, fp32 accumulation.
// GEMM tiles staged into LDS by the Tensor Data Mover (tensor_load_to_lds,
// TENSORcnt-synchronized, double-buffered) when the builtin is available.
// B=32 H=W=56 C=192 NH=6 WS=7 SS=3 -> N=49, NW=64, HEAD_D=32, HID=768
// M (tokens) = 32*3136 = 100352 = 784*128
// ---------------------------------------------------------------------------

typedef __attribute__((ext_vector_type(16))) _Float16 v16h;
typedef __attribute__((ext_vector_type(8)))  float    v8f;

struct alignas(16) H8s { _Float16 h[8]; };
struct alignas(16) I4s { int x, y, z, w; };

#define MTOK 100352

#if __has_builtin(__builtin_amdgcn_tensor_load_to_lds)
#define USE_TDM 1
#else
#define USE_TDM 0
#endif

__device__ __forceinline__ v8f wmma16(v16h a, v16h b, v8f c) {
  return __builtin_amdgcn_wmma_f32_16x16x32_f16(false, a, false, b, (short)0, c,
                                                false, false);
}

// 16x32 f16 fragment load from an LDS tile stored "major-index x 32 contiguous
// K values" with row pitch `ld` halfs.  Per CDNA5 ISA 16-bit A layout:
// lane L -> row (L&15); elements 0..7 -> K = 8*(L>>4)+e, elements 8..15 ->
// K = 16+8*(L>>4)+(e-8).  B fragments share the pattern for N-major tiles.
__device__ __forceinline__ v16h load_frag(const _Float16* p0, int row, int ld,
                                          int k0) {
  const int lane = threadIdx.x & 31;
  const int hf = lane >> 4;
  const _Float16* p = p0 + (size_t)(row + (lane & 15)) * ld + k0 + hf * 8;
  H8s lo = *(const H8s*)p;
  H8s hi = *(const H8s*)(p + 16);
  v16h f;
#pragma unroll
  for (int i = 0; i < 8; ++i) { f[i] = lo.h[i]; f[i + 8] = hi.h[i]; }
  return f;
}

#if USE_TDM
typedef unsigned int v4u  __attribute__((ext_vector_type(4)));
typedef int          v8ii __attribute__((ext_vector_type(8)));
typedef int          v4ii __attribute__((ext_vector_type(4)));

// TDM 2D tile load: tile_rows x 32 f16 elements from a row-major [rows_tensor
// x row_len] f16 tensor into LDS, padding +4 dwords per 16 dwords so the LDS
// row pitch is 40 halfs (matches load_frag).  D# per CDNA5 ISA ch.8:
//   group0: count=1 | lds_addr | global_addr[56:0] | type=2
//   group1: data_size=2B, pad_enable, pad_interval=16dw, pad_amount=4dw,
//           tensor_dim0/1, tile_dim0=32, tile_dim1=tile_rows, dim0_stride
__device__ __forceinline__ void tdm_load_tile_f16(const _Float16* gsrc,
                                                  const _Float16* lds_dst,
                                                  unsigned rows_tensor,
                                                  unsigned row_len,
                                                  unsigned tile_rows) {
  unsigned long long ga = (unsigned long long)(uintptr_t)gsrc;
  v4u g0;
  g0[0] = 1u;                                   // count=1 (valid user D#)
  g0[1] = (unsigned)(uintptr_t)lds_dst;         // LDS byte address
  g0[2] = (unsigned)ga;                         // global_addr[31:0]
  g0[3] = (unsigned)((ga >> 32) & 0x1FFFFFFu) | (2u << 30);  // [56:32]|type=2
  v8ii g1;
  g1[0] = (int)((1u << 16) |                    // data_size = 2 bytes
                (1u << 20) |                    // pad_enable
                (3u << 22) |                    // pad_interval: 16 dwords
                (3u << 25));                    // pad_amount:   4 dwords
  g1[1] = (int)((row_len & 0xFFFFu) << 16);     // tensor_dim0[15:0]
  g1[2] = (int)(((row_len >> 16) & 0xFFFFu) | ((rows_tensor & 0xFFFFu) << 16));
  g1[3] = (int)(((rows_tensor >> 16) & 0xFFFFu) | (32u << 16));  // tile_dim0=32
  g1[4] = (int)(tile_rows & 0xFFFFu);           // tile_dim1 ; tile_dim2=0
  g1[5] = (int)row_len;                         // tensor_dim0_stride[31:0]
  g1[6] = 0;
  g1[7] = 0;
  v4ii z4 = {0, 0, 0, 0};
#if __clang_major__ >= 23
  v8ii z8 = {0, 0, 0, 0, 0, 0, 0, 0};
  __builtin_amdgcn_tensor_load_to_lds(g0, g1, z4, z4, z8, 0);
#else
  __builtin_amdgcn_tensor_load_to_lds(g0, g1, z4, z4, 0);
#endif
}
#endif  // USE_TDM

// ---------------------------------------------------------------------------
// weights fp32 -> f16 (once per launch; tiny)
// ---------------------------------------------------------------------------
__global__ __launch_bounds__(256) void cvt_kernel(const float* __restrict__ s,
                                                  _Float16* __restrict__ d,
                                                  int n) {
  const int i = blockIdx.x * 256 + threadIdx.x;
  if (i < n) d[i] = (_Float16)s[i];
}

// ---------------------------------------------------------------------------
// LayerNorm (+ optional cyclic-shift + window-partition gather) -> f16
// one wave per token; 192 channels = 6 per lane
// ---------------------------------------------------------------------------
__global__ __launch_bounds__(256) void ln_kernel(
    const float* __restrict__ x, const float* __restrict__ g,
    const float* __restrict__ bb, _Float16* __restrict__ o, int shifted) {
  const int token = blockIdx.x * 8 + (threadIdx.x >> 5);
  const int lane = threadIdx.x & 31;
  size_t srow;
  if (shifted) {
    const int wwin = token / 49, n = token - wwin * 49;
    const int bimg = wwin >> 6, wi = wwin & 63;
    const int wr = wi >> 3, wc = wi & 7;
    const int th = n / 7, tw = n - th * 7;
    int hh = wr * 7 + th + 3; if (hh >= 56) hh -= 56;
    int wl = wc * 7 + tw + 3; if (wl >= 56) wl -= 56;
    srow = (size_t)bimg * 3136 + hh * 56 + wl;
  } else {
    srow = (size_t)token;
  }
  const float* px = x + srow * 192;
  float v[6];
  float s = 0.0f;
#pragma unroll
  for (int i = 0; i < 6; ++i) { v[i] = px[lane + 32 * i]; s += v[i]; }
#pragma unroll
  for (int d = 1; d < 32; d <<= 1) s += __shfl_xor(s, d);
  const float mean = s * (1.0f / 192.0f);
  float q = 0.0f;
#pragma unroll
  for (int i = 0; i < 6; ++i) { const float dd = v[i] - mean; q += dd * dd; }
#pragma unroll
  for (int d = 1; d < 32; d <<= 1) q += __shfl_xor(q, d);
  const float rstd = rsqrtf(q * (1.0f / 192.0f) + 1e-5f);
  _Float16* po = o + (size_t)token * 192;
#pragma unroll
  for (int i = 0; i < 6; ++i) {
    const int c = lane + 32 * i;
    po[c] = (_Float16)((v[i] - mean) * rstd * g[c] + bb[c]);
  }
}

// ---------------------------------------------------------------------------
// Tiled WMMA GEMM:  out[M,N] = A[M,K](f16) * W[N,K]^T(f16) + bias(f32)
// block tile 128x64, 8 waves (4x2), each wave 32x32 (4 wmma / k-step).
// LDS double-buffered; tiles staged by TDM (tensor_load_to_lds) when
// available, else by vector copies.
// MODE 0: store f16           MODE 1: exact GELU, store f16
// MODE 2: swin reverse+roll scatter, outF = resid + v (fp32)
// MODE 3: outF += v (fp32, residual accumulate)
// ---------------------------------------------------------------------------
template <int MODE>
__global__ __launch_bounds__(256) void gemm_wmma(
    const _Float16* __restrict__ A, const _Float16* __restrict__ W,
    const float* __restrict__ bias, int M, int N, int K,
    _Float16* __restrict__ outH, float* __restrict__ outF,
    const float* __restrict__ resid) {
  __shared__ _Float16 sA[2][128 * 40];
  __shared__ _Float16 sB[2][64 * 40];

  const int t = threadIdx.x;
  const int wave = t >> 5;
  const int mw = wave & 3;
  const int nw = wave >> 2;
  const int m0 = blockIdx.x * 128;
  const int n0 = blockIdx.y * 64;
  const int nk = K >> 5;

#if USE_TDM
  if (wave == 0) {
    tdm_load_tile_f16(A + (size_t)m0 * K, sA[0], (unsigned)M, (unsigned)K, 128);
    tdm_load_tile_f16(W + (size_t)n0 * K, sB[0], (unsigned)N, (unsigned)K, 64);
  }
#else
  {  // manual stage of k-tile 0 into buffer 0
    const int r = t >> 1;
    const int c = (t & 1) * 16;
    const _Float16* gp = A + (size_t)(m0 + r) * K + c;
    I4s a0 = *(const I4s*)gp;
    I4s a1 = *(const I4s*)(gp + 8);
    *(I4s*)(sA[0] + r * 40 + c) = a0;
    *(I4s*)(sA[0] + r * 40 + c + 8) = a1;
    if (t < 128) {
      const _Float16* gw = W + (size_t)(n0 + r) * K + c;
      I4s b0 = *(const I4s*)gw;
      I4s b1 = *(const I4s*)(gw + 8);
      *(I4s*)(sB[0] + r * 40 + c) = b0;
      *(I4s*)(sB[0] + r * 40 + c + 8) = b1;
    }
  }
#endif

  v8f acc[2][2] = {};
  for (int kt = 0; kt < nk; ++kt) {
    const int cur = kt & 1;
#if USE_TDM
    if (wave == 0) __builtin_amdgcn_s_wait_tensorcnt(0);
    __syncthreads();  // tile kt landed; all waves done with buffer cur^1
    if (kt + 1 < nk && wave == 0) {
      tdm_load_tile_f16(A + (size_t)m0 * K + (kt + 1) * 32, sA[cur ^ 1],
                        (unsigned)M, (unsigned)K, 128);
      tdm_load_tile_f16(W + (size_t)n0 * K + (kt + 1) * 32, sB[cur ^ 1],
                        (unsigned)N, (unsigned)K, 64);
    }
#else
    __syncthreads();  // stage(kt) visible; compute(kt-1) reads complete
    if (kt + 1 < nk) {
      const int r = t >> 1;
      const int c = (t & 1) * 16;
      const _Float16* gp = A + (size_t)(m0 + r) * K + (kt + 1) * 32 + c;
      I4s a0 = *(const I4s*)gp;
      I4s a1 = *(const I4s*)(gp + 8);
      *(I4s*)(sA[cur ^ 1] + r * 40 + c) = a0;
      *(I4s*)(sA[cur ^ 1] + r * 40 + c + 8) = a1;
      if (t < 128) {
        const _Float16* gw = W + (size_t)(n0 + r) * K + (kt + 1) * 32 + c;
        I4s b0 = *(const I4s*)gw;
        I4s b1 = *(const I4s*)(gw + 8);
        *(I4s*)(sB[cur ^ 1] + r * 40 + c) = b0;
        *(I4s*)(sB[cur ^ 1] + r * 40 + c + 8) = b1;
      }
    }
#endif
    v16h a0 = load_frag(sA[cur], mw * 32, 40, 0);
    v16h a1 = load_frag(sA[cur], mw * 32 + 16, 40, 0);
    v16h b0 = load_frag(sB[cur], nw * 32, 40, 0);
    v16h b1 = load_frag(sB[cur], nw * 32 + 16, 40, 0);
    acc[0][0] = wmma16(a0, b0, acc[0][0]);
    acc[0][1] = wmma16(a0, b1, acc[0][1]);
    acc[1][0] = wmma16(a1, b0, acc[1][0]);
    acc[1][1] = wmma16(a1, b1, acc[1][1]);
  }

  const int lane = t & 31;
  const int ln = lane & 15, hf = lane >> 4;
#pragma unroll
  for (int im = 0; im < 2; ++im) {
#pragma unroll
    for (int in_ = 0; in_ < 2; ++in_) {
      const int gn = n0 + nw * 32 + in_ * 16 + ln;
      const float bv = bias[gn];
#pragma unroll
      for (int j = 0; j < 8; ++j) {
        const int gm = m0 + mw * 32 + im * 16 + j + 8 * hf;
        float v = acc[im][in_][j] + bv;
        if (MODE == 0) {
          outH[(size_t)gm * N + gn] = (_Float16)v;
        } else if (MODE == 1) {
          v = 0.5f * v * (1.0f + erff(v * 0.70710678f));
          outH[(size_t)gm * N + gn] = (_Float16)v;
        } else if (MODE == 2) {
          const int wwin = gm / 49, n_ = gm - wwin * 49;
          const int bimg = wwin >> 6, wi = wwin & 63;
          const int wr = wi >> 3, wc = wi & 7;
          const int th = n_ / 7, tw = n_ - th * 7;
          int hh = wr * 7 + th + 3; if (hh >= 56) hh -= 56;
          int wl = wc * 7 + tw + 3; if (wl >= 56) wl -= 56;
          const size_t idx = ((size_t)bimg * 3136 + hh * 56 + wl) * 192 + gn;
          outF[idx] = resid[idx] + v;
        } else {
          const size_t idx = (size_t)gm * N + gn;
          outF[idx] = outF[idx] + v;
        }
      }
    }
  }
}

// ---------------------------------------------------------------------------
// Windowed attention: one wave per (window, head).
// S = QK^T (16 wmma), softmax(+rel-pos bias, +shift mask) in regs, O = PV
// (16 wmma).  Tokens padded 49 -> 64 (K/V pad rows zeroed; pad cols -> -1e30).
// ---------------------------------------------------------------------------
__global__ __launch_bounds__(32) void attn_kernel(
    const _Float16* __restrict__ qkv, const float* __restrict__ btab,
    _Float16* __restrict__ outA) {
  __shared__ _Float16 sQ[64 * 40];
  __shared__ _Float16 sK[64 * 40];
  __shared__ _Float16 sVT[32 * 72];
  __shared__ _Float16 sP[64 * 72];

  const int wh = blockIdx.x;
  const int w = wh / 6, h = wh - w * 6;
  const int wi = w & 63;
  const int wr = wi >> 3, wc = wi & 7;
  const int lane = threadIdx.x;
  const int ln = lane & 15, hf = lane >> 4;

  const _Float16* base = qkv + (size_t)w * 49 * 576 + h * 32;
  for (int n = lane; n < 64; n += 32) {
    H8s* dq = (H8s*)(sQ + n * 40);
    H8s* dk = (H8s*)(sK + n * 40);
    if (n < 49) {
      const H8s* sq = (const H8s*)(base + (size_t)n * 576);
      const H8s* sk = (const H8s*)(base + (size_t)n * 576 + 192);
      const H8s* sv = (const H8s*)(base + (size_t)n * 576 + 384);
      H8s v0 = sv[0], v1 = sv[1], v2 = sv[2], v3 = sv[3];
      dq[0] = sq[0]; dq[1] = sq[1]; dq[2] = sq[2]; dq[3] = sq[3];
      dk[0] = sk[0]; dk[1] = sk[1]; dk[2] = sk[2]; dk[3] = sk[3];
#pragma unroll
      for (int d = 0; d < 8; ++d) {
        sVT[(d) * 72 + n] = v0.h[d];
        sVT[(d + 8) * 72 + n] = v1.h[d];
        sVT[(d + 16) * 72 + n] = v2.h[d];
        sVT[(d + 24) * 72 + n] = v3.h[d];
      }
    } else {
      H8s z = {};
      dq[0] = z; dq[1] = z; dq[2] = z; dq[3] = z;
      dk[0] = z; dk[1] = z; dk[2] = z; dk[3] = z;
#pragma unroll
      for (int d = 0; d < 32; ++d) sVT[d * 72 + n] = (_Float16)0.0f;
    }
  }
  __syncthreads();

  // S = Q K^T  (K stored row-major [token][d] == N-major for the B fragment)
  v16h aq[4];
#pragma unroll
  for (int ti = 0; ti < 4; ++ti) aq[ti] = load_frag(sQ, ti * 16, 40, 0);
  v8f S[4][4] = {};
#pragma unroll
  for (int tj = 0; tj < 4; ++tj) {
    v16h bk = load_frag(sK, tj * 16, 40, 0);
#pragma unroll
    for (int ti = 0; ti < 4; ++ti) S[ti][tj] = wmma16(aq[ti], bk, S[ti][tj]);
  }

  // per-lane column metadata (4 key columns per lane)
  int colJ[4], thJ[4], twJ[4], regJ[4], okJ[4];
#pragma unroll
  for (int tj = 0; tj < 4; ++tj) {
    const int cj = tj * 16 + ln;
    colJ[tj] = cj;
    okJ[tj] = (cj < 49);
    const int tjh = cj / 7, tjw = cj - tjh * 7;
    thJ[tj] = tjh; twJ[tj] = tjw;
    const int hj = wr * 7 + tjh, wj = wc * 7 + tjw;
    regJ[tj] = (hj < 49 ? 0 : (hj < 53 ? 1 : 2)) * 3 +
               (wj < 49 ? 0 : (wj < 53 ? 1 : 2));
  }

  const float scale = 0.17677669529663687f;  // 1/sqrt(32)
#pragma unroll
  for (int ti = 0; ti < 4; ++ti) {
#pragma unroll
    for (int j = 0; j < 8; ++j) {
      const int m = ti * 16 + j + 8 * hf;  // uniform across each 16-lane group
      if (m < 49) {
        const int th = m / 7, tw = m - th * 7;
        const int hi_ = wr * 7 + th, wi_ = wc * 7 + tw;
        const int regi = (hi_ < 49 ? 0 : (hi_ < 53 ? 1 : 2)) * 3 +
                         (wi_ < 49 ? 0 : (wi_ < 53 ? 1 : 2));
        float vals[4];
#pragma unroll
        for (int tj = 0; tj < 4; ++tj) {
          if (okJ[tj]) {
            const int rel = (th - thJ[tj] + 6) * 13 + (tw - twJ[tj] + 6);
            float vv = S[ti][tj][j] * scale + btab[rel * 6 + h];
            if (regi != regJ[tj]) vv -= 100.0f;
            vals[tj] = vv;
          } else {
            vals[tj] = -1e30f;
          }
        }
        float mx = fmaxf(fmaxf(vals[0], vals[1]), fmaxf(vals[2], vals[3]));
#pragma unroll
        for (int d = 1; d < 16; d <<= 1) mx = fmaxf(mx, __shfl_xor(mx, d));
        float sum = 0.0f;
#pragma unroll
        for (int tj = 0; tj < 4; ++tj) {
          vals[tj] = __expf(vals[tj] - mx);
          sum += vals[tj];
        }
#pragma unroll
        for (int d = 1; d < 16; d <<= 1) sum += __shfl_xor(sum, d);
        const float inv = 1.0f / sum;
#pragma unroll
        for (int tj = 0; tj < 4; ++tj)
          sP[m * 72 + colJ[tj]] = (_Float16)(vals[tj] * inv);
      }
    }
  }
  __syncthreads();

  // O = P V   (V^T stored [d][token] == N-major for the B fragment)
  v8f O[4][2] = {};
#pragma unroll
  for (int kt = 0; kt < 2; ++kt) {
    v16h ap[4];
#pragma unroll
    for (int ti = 0; ti < 4; ++ti) ap[ti] = load_frag(sP, ti * 16, 72, kt * 32);
#pragma unroll
    for (int tn = 0; tn < 2; ++tn) {
      v16h bv = load_frag(sVT, tn * 16, 72, kt * 32);
#pragma unroll
      for (int ti = 0; ti < 4; ++ti) O[ti][tn] = wmma16(ap[ti], bv, O[ti][tn]);
    }
  }
#pragma unroll
  for (int ti = 0; ti < 4; ++ti) {
#pragma unroll
    for (int tn = 0; tn < 2; ++tn) {
#pragma unroll
      for (int j = 0; j < 8; ++j) {
        const int m = ti * 16 + j + 8 * hf;
        if (m < 49) {
          const int d = tn * 16 + ln;
          outA[((size_t)w * 49 + m) * 192 + h * 32 + d] = (_Float16)O[ti][tn][j];
        }
      }
    }
  }
}

// ---------------------------------------------------------------------------
extern "C" void kernel_launch(void* const* d_in, const int* in_sizes, int n_in,
                              void* d_out, int out_size, void* d_ws,
                              size_t ws_size, hipStream_t stream) {
  (void)in_sizes; (void)n_in; (void)out_size; (void)ws_size;
  const float* x      = (const float*)d_in[0];
  const float* g1     = (const float*)d_in[1];
  const float* b1     = (const float*)d_in[2];
  const float* qkv_w  = (const float*)d_in[3];
  const float* qkv_b  = (const float*)d_in[4];
  const float* proj_w = (const float*)d_in[5];
  const float* proj_b = (const float*)d_in[6];
  const float* bias_t = (const float*)d_in[7];
  const float* g2     = (const float*)d_in[8];
  const float* b2     = (const float*)d_in[9];
  const float* fc1_w  = (const float*)d_in[10];
  const float* fc1_b  = (const float*)d_in[11];
  const float* fc2_w  = (const float*)d_in[12];
  const float* fc2_b  = (const float*)d_in[13];
  float* out = (float*)d_out;

  char* ws = (char*)d_ws;
  size_t off = 0;
  auto carve = [&](size_t bytes) {
    char* p = ws + off;
    off += (bytes + 255) & ~(size_t)255;
    return p;
  };
  _Float16* bufXW  = (_Float16*)carve((size_t)MTOK * 192 * 2);  // LN1+shifted
  _Float16* bufQKV = (_Float16*)carve((size_t)MTOK * 576 * 2);
  _Float16* bufATT = (_Float16*)carve((size_t)MTOK * 192 * 2);
  _Float16* bufLN2 = (_Float16*)carve((size_t)MTOK * 192 * 2);
  _Float16* bufHID = (_Float16*)carve((size_t)MTOK * 768 * 2);
  _Float16* wQKV   = (_Float16*)carve((size_t)576 * 192 * 2);
  _Float16* wPROJ  = (_Float16*)carve((size_t)192 * 192 * 2);
  _Float16* wFC1   = (_Float16*)carve((size_t)768 * 192 * 2);
  _Float16* wFC2   = (_Float16*)carve((size_t)192 * 768 * 2);

  // 0) weights -> f16 (tiny)
  cvt_kernel<<<(576 * 192 + 255) / 256, 256, 0, stream>>>(qkv_w, wQKV, 576 * 192);
  cvt_kernel<<<(192 * 192 + 255) / 256, 256, 0, stream>>>(proj_w, wPROJ, 192 * 192);
  cvt_kernel<<<(768 * 192 + 255) / 256, 256, 0, stream>>>(fc1_w, wFC1, 768 * 192);
  cvt_kernel<<<(192 * 768 + 255) / 256, 256, 0, stream>>>(fc2_w, wFC2, 192 * 768);

  // 1) LN1 + cyclic shift + window partition (f16)
  ln_kernel<<<MTOK / 8, 256, 0, stream>>>(x, g1, b1, bufXW, 1);
  // 2) QKV GEMM: [100352,192] x [192,576]
  gemm_wmma<0><<<dim3(MTOK / 128, 576 / 64), 256, 0, stream>>>(
      bufXW, wQKV, qkv_b, MTOK, 576, 192, bufQKV, nullptr, nullptr);
  // 3) windowed attention (2048 windows x 6 heads)
  attn_kernel<<<2048 * 6, 32, 0, stream>>>(bufQKV, bias_t, bufATT);
  // 4) proj GEMM + reverse/roll scatter + residual -> d_out (fp32)
  gemm_wmma<2><<<dim3(MTOK / 128, 192 / 64), 256, 0, stream>>>(
      bufATT, wPROJ, proj_b, MTOK, 192, 192, nullptr, out, x);
  // 5) LN2 (f16)
  ln_kernel<<<MTOK / 8, 256, 0, stream>>>(out, g2, b2, bufLN2, 0);
  // 6) FC1 GEMM + exact GELU (f16)
  gemm_wmma<1><<<dim3(MTOK / 128, 768 / 64), 256, 0, stream>>>(
      bufLN2, wFC1, fc1_b, MTOK, 768, 192, bufHID, nullptr, nullptr);
  // 7) FC2 GEMM + residual accumulate into d_out
  gemm_wmma<3><<<dim3(MTOK / 128, 192 / 64), 256, 0, stream>>>(
      bufHID, wFC2, fc2_b, MTOK, 192, 768, nullptr, out, nullptr);
}
